// GlobalLocalAttention_49271864820180
// MI455X (gfx1250) — compile-verified, matrix operand order checked by reading, not run on hardware
//
#include <hip/hip_runtime.h>
#include <hip/hip_bf16.h>

typedef __attribute__((ext_vector_type(16))) __bf16 v16bf;
typedef __attribute__((ext_vector_type(8)))  __bf16 v8bf;
typedef __attribute__((ext_vector_type(8)))  float  v8f;
typedef __attribute__((ext_vector_type(4)))  unsigned int u32x4;

#define BDIM  4
#define SDIM  4096
#define HDIM  1024
#define PDIM  4096
#define DDIM  512
#define MTOK  (BDIM*SDIM)     // 16384 tokens
#define H3    (3*HDIM)        // 3072

// ---- GEMM tiling: block 128x128x32, 8 waves, wave tile 32x64 (2x4 wmma) ----
#define BM 128
#define BN 128
#define BK 32
#define LDT 40                // BK + 8 halves pad (16B) to dodge LDS bank conflicts
#define GEMM_ACC  1
#define GEMM_RELU 2

// C = A[M,K](bf16) * W[N,K](bf16)^T (+bias[N]) (+=C) (relu), C is f32 with row stride ldc.
// shift != 0: A row r reads row r+shift, zero outside the row's S=4096 segment (dilated conv taps).
// Double-buffered LDS staging via CDNA5 async global->LDS DMA (ASYNCcnt).
__global__ __launch_bounds__(256) void gemm_bf16_wmma(
    const __bf16* __restrict__ A, const __bf16* __restrict__ W,
    const float* __restrict__ bias, float* __restrict__ C,
    int M, int N, int K, int ldc, int shift, int flags)
{
  __shared__ __bf16 smem[2][(BM + BN) * LDT];   // 40KB: [buf][A(128x40) | B(128x40)]

  const int tid  = threadIdx.x;
  const int lane = tid & 31;
  const int wid  = tid >> 5;
  const int wm   = wid & 3;     // 4 waves along M (32 rows each)
  const int wn   = wid >> 2;    // 2 waves along N (64 cols each)
  const int blockM = blockIdx.y * BM;
  const int blockN = blockIdx.x * BN;
  const int r = lane & 15;
  const int g = lane >> 4;

  // ---- per-thread staging chunks: c in {tid, tid+256}; chunk = 16B of one tile row ----
  int srcA[2];
  const __bf16* pA[2];
  const __bf16* pB[2];
  unsigned offA[2][2], offB[2][2];
#pragma unroll
  for (int i = 0; i < 2; ++i) {
    const int c    = tid + i * 256;
    const int row  = c >> 2;
    const int col8 = (c & 3) << 3;
    const int gRow = blockM + row;
    int src = gRow;
    if (shift != 0) {
      int s = gRow + shift;
      src = (s >= 0 && s < M && ((s >> 12) == (gRow >> 12))) ? s : -1;
    }
    srcA[i] = src;
    pA[i] = A + (size_t)(src < 0 ? 0 : src) * K + col8;
    pB[i] = W + (size_t)(blockN + row) * K + col8;
#pragma unroll
    for (int b = 0; b < 2; ++b) {
      offA[i][b] = (unsigned)(uintptr_t)&smem[b][row * LDT + col8];
      offB[i][b] = (unsigned)(uintptr_t)&smem[b][BM * LDT + row * LDT + col8];
    }
    if (src < 0) {  // zero once; these chunks are never targeted by async loads
      u32x4 z; z[0] = 0u; z[1] = 0u; z[2] = 0u; z[3] = 0u;
      *(u32x4*)&smem[0][row * LDT + col8] = z;
      *(u32x4*)&smem[1][row * LDT + col8] = z;
    }
  }

  v8f acc[2][4];
#pragma unroll
  for (int i = 0; i < 2; ++i)
#pragma unroll
    for (int j = 0; j < 4; ++j)
#pragma unroll
      for (int v = 0; v < 8; ++v) acc[i][j][v] = 0.0f;

  const int nk = K / BK;

  // async global->LDS stage of K-slab kt into buffer b (4 x 16B per thread)
  auto stage = [&](int kt, int b) {
    const int kof = kt * BK;
#pragma unroll
    for (int i = 0; i < 2; ++i) {
      if (srcA[i] >= 0) {
        unsigned long long ga = (unsigned long long)(uintptr_t)(pA[i] + kof);
        asm volatile("global_load_async_to_lds_b128 %0, %1, off"
                     :: "v"(offA[i][b]), "v"(ga) : "memory");
      }
      unsigned long long gb = (unsigned long long)(uintptr_t)(pB[i] + kof);
      asm volatile("global_load_async_to_lds_b128 %0, %1, off"
                   :: "v"(offB[i][b]), "v"(gb) : "memory");
    }
  };

  stage(0, 0);
  asm volatile("s_wait_asynccnt 0x0" ::: "memory");
  __syncthreads();

  for (int kt = 0; kt < nk; ++kt) {
    const int buf = kt & 1;
    if (kt + 1 < nk) stage(kt + 1, buf ^ 1);   // DMA next slab while we compute

    const __bf16* sA = &smem[buf][0];
    const __bf16* sB = &smem[buf][BM * LDT];

    // fragments: 16-bit A-matrix 16x32 layout -> two 16B LDS reads per lane
    v16bf aF[2], bF[4];
#pragma unroll
    for (int tm = 0; tm < 2; ++tm) {
      const __bf16* p = &sA[(wm * 32 + tm * 16 + r) * LDT + 8 * g];
      union { v16bf v; v8bf h[2]; } u;
      u.h[0] = *(const v8bf*)(p);
      u.h[1] = *(const v8bf*)(p + 16);
      aF[tm] = u.v;
    }
#pragma unroll
    for (int tn = 0; tn < 4; ++tn) {
      const __bf16* p = &sB[(wn * 64 + tn * 16 + r) * LDT + 8 * g];
      union { v16bf v; v8bf h[2]; } u;
      u.h[0] = *(const v8bf*)(p);
      u.h[1] = *(const v8bf*)(p + 16);
      bF[tn] = u.v;
    }
#pragma unroll
    for (int tm = 0; tm < 2; ++tm)
#pragma unroll
      for (int tn = 0; tn < 4; ++tn)
        acc[tm][tn] = __builtin_amdgcn_wmma_f32_16x16x32_bf16(
            false, aF[tm], false, bF[tn], (short)0, acc[tm][tn], false, false);

    if (kt + 1 < nk) {
      asm volatile("s_wait_asynccnt 0x0" ::: "memory");  // next slab landed
      __syncthreads();                                   // everyone done reading buf
    }
  }

  // epilogue: C/D layout: lane -> n = lane&15, m = v + 8*(lane>>4)
#pragma unroll
  for (int tm = 0; tm < 2; ++tm) {
#pragma unroll
    for (int tn = 0; tn < 4; ++tn) {
      int gN = blockN + wn * 64 + tn * 16 + r;
      float bv = bias ? bias[gN] : 0.0f;
#pragma unroll
      for (int v = 0; v < 8; ++v) {
        int gM = blockM + wm * 32 + tm * 16 + v + 8 * g;
        size_t idx = (size_t)gM * ldc + gN;
        float o = acc[tm][tn][v] + bv;
        if (flags & GEMM_ACC)  o += C[idx];
        if (flags & GEMM_RELU) o = fmaxf(o, 0.0f);
        C[idx] = o;
      }
    }
  }
}

// ---------------- elementwise / conversion kernels ----------------
__global__ void k_f32_to_bf16(const float* __restrict__ in, __bf16* __restrict__ out, size_t n) {
  size_t i = (size_t)blockIdx.x * blockDim.x + threadIdx.x;
  size_t st = (size_t)gridDim.x * blockDim.x;
  for (; i < n; i += st) out[i] = (__bf16)in[i];
}

// conv_w [d][oc][ic][kk] -> bf16 [d][kk][oc][ic] (K-contiguous weight rows per tap)
__global__ void k_convw_reorg(const float* __restrict__ cw, __bf16* __restrict__ out) {
  const size_t n = (size_t)9 * HDIM * HDIM;
  size_t i = (size_t)blockIdx.x * blockDim.x + threadIdx.x;
  size_t st = (size_t)gridDim.x * blockDim.x;
  for (; i < n; i += st) {
    int ic = (int)(i % HDIM);
    int oc = (int)((i / HDIM) % HDIM);
    int kk = (int)((i / ((size_t)HDIM * HDIM)) % 3);
    int d  = (int)(i / ((size_t)3 * HDIM * HDIM));
    out[i] = (__bf16)cw[(((size_t)d * HDIM + oc) * HDIM + ic) * 3 + kk];
  }
}

__global__ void k_gelu_bf16(const float* __restrict__ in, __bf16* __restrict__ out, size_t n) {
  size_t i = (size_t)blockIdx.x * blockDim.x + threadIdx.x;
  size_t st = (size_t)gridDim.x * blockDim.x;
  for (; i < n; i += st) {
    float x = in[i];
    out[i] = (__bf16)(0.5f * x * (1.0f + erff(x * 0.70710678118654752f)));
  }
}

__global__ void k_zero(float* __restrict__ p, int n) {
  int i = blockIdx.x * blockDim.x + threadIdx.x;
  if (i < n) p[i] = 0.0f;
}

// ---------------- router: top-2 + softmax + expert mix + usage ----------------
__global__ __launch_bounds__(256) void k_router(
    const float* __restrict__ logits, const float* __restrict__ pool,
    const float* __restrict__ temp_p, float* __restrict__ wmapvec,
    float* __restrict__ usage)
{
  __shared__ float sv[256];
  __shared__ int   si[256];
  __shared__ float s_v1, s_w1, s_w2;
  __shared__ int   s_i1, s_i2;
  const int t = blockIdx.x, tid = threadIdx.x;
  const float* row = logits + (size_t)t * PDIM;
  float temp = fminf(fmaxf(temp_p[0], 0.1f), 5.0f);
  const float inv = 1.0f / temp;

  // pass 1: argmax of clipped scaled logits (lowest index wins ties, like lax.top_k)
  float bv = -3.4e38f; int bi = PDIM;
  for (int p = tid; p < PDIM; p += 256) {
    float v = fminf(fmaxf(row[p] * inv, -10.0f), 10.0f);
    if (v > bv || (v == bv && p < bi)) { bv = v; bi = p; }
  }
  sv[tid] = bv; si[tid] = bi; __syncthreads();
  for (int s = 128; s > 0; s >>= 1) {
    if (tid < s) {
      float ov = sv[tid + s]; int oi = si[tid + s];
      if (ov > sv[tid] || (ov == sv[tid] && oi < si[tid])) { sv[tid] = ov; si[tid] = oi; }
    }
    __syncthreads();
  }
  if (tid == 0) { s_v1 = sv[0]; s_i1 = si[0]; }
  __syncthreads();
  const int i1 = s_i1; const float v1 = s_v1;

  // pass 2: argmax excluding i1
  bv = -3.4e38f; bi = PDIM;
  for (int p = tid; p < PDIM; p += 256) {
    if (p == i1) continue;
    float v = fminf(fmaxf(row[p] * inv, -10.0f), 10.0f);
    if (v > bv || (v == bv && p < bi)) { bv = v; bi = p; }
  }
  sv[tid] = bv; si[tid] = bi; __syncthreads();
  for (int s = 128; s > 0; s >>= 1) {
    if (tid < s) {
      float ov = sv[tid + s]; int oi = si[tid + s];
      if (ov > sv[tid] || (ov == sv[tid] && oi < si[tid])) { sv[tid] = ov; si[tid] = oi; }
    }
    __syncthreads();
  }
  if (tid == 0) {
    float v2 = sv[0];
    float w1 = 1.0f / (1.0f + expf(v2 - v1));
    s_i2 = si[0]; s_w1 = w1; s_w2 = 1.0f - w1;
    atomicAdd(&usage[i1], w1);
    atomicAdd(&usage[si[0]], 1.0f - w1);
  }
  __syncthreads();
  const float w1 = s_w1, w2 = s_w2;
  const float* p1 = pool + (size_t)i1 * DDIM;
  const float* p2 = pool + (size_t)s_i2 * DDIM;
  float* wm = wmapvec + (size_t)t * DDIM;
  for (int j = tid; j < DDIM; j += 256) wm[j] = w1 * p1[j] + w2 * p2[j];
}

// ---------------- layernorm(proj) + pack [ln(proj), weighted_map] as bf16 ----------------
__global__ __launch_bounds__(256) void k_ln_pack(
    const float* __restrict__ proj, const float* __restrict__ wmapvec,
    const float* __restrict__ gw, const float* __restrict__ bw,
    __bf16* __restrict__ comb)
{
  __shared__ float red[256];
  __shared__ float s_mu, s_rs;
  const int t = blockIdx.x, tid = threadIdx.x;
  const float* x = proj + (size_t)t * DDIM;
  float x0 = x[tid], x1 = x[tid + 256];
  red[tid] = x0 + x1; __syncthreads();
  for (int s = 128; s > 0; s >>= 1) { if (tid < s) red[tid] += red[tid + s]; __syncthreads(); }
  if (tid == 0) s_mu = red[0] * (1.0f / DDIM);
  __syncthreads();
  float mu = s_mu, d0 = x0 - mu, d1 = x1 - mu;
  red[tid] = d0 * d0 + d1 * d1; __syncthreads();
  for (int s = 128; s > 0; s >>= 1) { if (tid < s) red[tid] += red[tid + s]; __syncthreads(); }
  if (tid == 0) s_rs = rsqrtf(red[0] * (1.0f / DDIM) + 1e-5f);
  __syncthreads();
  float rs = s_rs;
  __bf16* c = comb + (size_t)t * (2 * DDIM);
  c[tid]       = (__bf16)(d0 * rs * gw[tid] + bw[tid]);
  c[tid + 256] = (__bf16)(d1 * rs * gw[tid + 256] + bw[tid + 256]);
  const float* wmv = wmapvec + (size_t)t * DDIM;
  c[DDIM + tid]       = (__bf16)wmv[tid];
  c[DDIM + tid + 256] = (__bf16)wmv[tid + 256];
}

// ---------------- gate = sigmoid(x.wg + bg); out = gate*local + (1-gate)*global ----------------
__global__ __launch_bounds__(256) void k_gate_combine(
    const float* __restrict__ x, const float* __restrict__ wg, const float* __restrict__ bg,
    const float* __restrict__ localo, const float* __restrict__ globalo,
    float* __restrict__ out)
{
  __shared__ float red[256];
  __shared__ float s_gate;
  const int t = blockIdx.x, tid = threadIdx.x;
  const float* xr = x + (size_t)t * HDIM;
  float a = 0.0f;
  for (int j = tid; j < HDIM; j += 256) a += xr[j] * wg[j];
  red[tid] = a; __syncthreads();
  for (int s = 128; s > 0; s >>= 1) { if (tid < s) red[tid] += red[tid + s]; __syncthreads(); }
  if (tid == 0) s_gate = 1.0f / (1.0f + expf(-(red[0] + bg[0])));
  __syncthreads();
  float gate = s_gate;
  for (int j = tid; j < HDIM; j += 256) {
    size_t idx = (size_t)t * HDIM + j;
    out[idx] = gate * localo[idx] + (1.0f - gate) * globalo[idx];
  }
}

// ---------------- diversity loss (scale = min(1, B*S*H/(P*k)) = 1) ----------------
__global__ __launch_bounds__(256) void k_divloss(const float* __restrict__ usage, float* __restrict__ out) {
  __shared__ float red[256];
  __shared__ float s_sum;
  const int tid = threadIdx.x;
  float a = 0.0f;
  for (int p = tid; p < PDIM; p += 256) a += usage[p];
  red[tid] = a; __syncthreads();
  for (int s = 128; s > 0; s >>= 1) { if (tid < s) red[tid] += red[tid + s]; __syncthreads(); }
  if (tid == 0) s_sum = red[0] + 1e-8f;
  __syncthreads();
  const float inv = 1.0f / s_sum, uni = 1.0f / PDIM;
  float q = 0.0f;
  for (int p = tid; p < PDIM; p += 256) { float f = usage[p] * inv - uni; q += f * f; }
  red[tid] = q; __syncthreads();
  for (int s = 128; s > 0; s >>= 1) { if (tid < s) red[tid] += red[tid + s]; __syncthreads(); }
  if (tid == 0) out[0] = (red[0] / PDIM) * 0.01f;
}

// ============================ host orchestration ============================
extern "C" void kernel_launch(void* const* d_in, const int* in_sizes, int n_in,
                              void* d_out, int out_size, void* d_ws, size_t ws_size,
                              hipStream_t stream) {
  (void)in_sizes; (void)n_in; (void)out_size; (void)ws_size;
  const float* x      = (const float*)d_in[0];
  const float* pool   = (const float*)d_in[1];
  const float* w1     = (const float*)d_in[2];
  const float* b1     = (const float*)d_in[3];
  const float* w2     = (const float*)d_in[4];
  const float* b2     = (const float*)d_in[5];
  const float* temp   = (const float*)d_in[6];
  const float* wproj  = (const float*)d_in[7];
  const float* bproj  = (const float*)d_in[8];
  const float* ln_g   = (const float*)d_in[9];
  const float* ln_b   = (const float*)d_in[10];
  const float* wmap   = (const float*)d_in[11];
  const float* bmap   = (const float*)d_in[12];
  const float* conv_w = (const float*)d_in[13];
  const float* conv_b = (const float*)d_in[14];
  const float* wout   = (const float*)d_in[15];
  const float* bout   = (const float*)d_in[16];
  const float* wrp    = (const float*)d_in[17];
  const float* brp    = (const float*)d_in[18];
  const float* wg     = (const float*)d_in[19];
  const float* bg     = (const float*)d_in[20];
  float* out = (float*)d_out;

  // ---- workspace bump allocator (256B aligned) ----
  char* ws = (char*)d_ws;
  size_t off = 0;
  auto take = [&](size_t bytes) -> char* {
    char* p = ws + off;
    off = (off + bytes + 255) & ~(size_t)255;
    return p;
  };
  __bf16* xbf      = (__bf16*)take((size_t)MTOK * HDIM * 2);
  __bf16* w1bf     = (__bf16*)take((size_t)HDIM * HDIM * 2);
  __bf16* w2bf     = (__bf16*)take((size_t)PDIM * HDIM * 2);
  __bf16* wprojbf  = (__bf16*)take((size_t)DDIM * HDIM * 2);
  __bf16* wmapbf   = (__bf16*)take((size_t)DDIM * 2 * DDIM * 2);
  __bf16* woutbf   = (__bf16*)take((size_t)HDIM * H3 * 2);
  __bf16* wrpbf    = (__bf16*)take((size_t)HDIM * DDIM * 2);
  __bf16* convwbf  = (__bf16*)take((size_t)9 * HDIM * HDIM * 2);
  __bf16* convact  = (__bf16*)take((size_t)MTOK * H3 * 2);
  __bf16* interbf  = (__bf16*)take((size_t)MTOK * HDIM * 2);
  float*  wmapvec  = (float*)take((size_t)MTOK * DDIM * 4);
  __bf16* combbf   = (__bf16*)take((size_t)MTOK * 2 * DDIM * 2);
  __bf16* transfbf = (__bf16*)take((size_t)MTOK * DDIM * 2);
  float*  usage    = (float*)take((size_t)PDIM * 4);
  // region A: inter(f32) then logits(f32) (inter consumed before logits written)
  char* bigA = take((size_t)MTOK * PDIM * 4);
  float* interf = (float*)bigA;
  float* logits = (float*)bigA;
  // region B: conv acc [M,3H] f32, later recycled for localout/proj/transf/globalout
  char* bigB = take((size_t)MTOK * H3 * 4);
  float* convacc  = (float*)bigB;
  float* localout = (float*)bigB;                                          // 64MB
  float* projf    = (float*)(bigB + (size_t)MTOK * HDIM * 4);              // 32MB
  float* transff  = (float*)(bigB + (size_t)MTOK * (HDIM + DDIM) * 4);     // 32MB
  float* globalo  = (float*)(bigB + (size_t)MTOK * (HDIM + 2 * DDIM) * 4); // 64MB

  const dim3 cb(256);
  const int cgrid = 4096;

  // 0) zero usage (ws is not re-poisoned between replays; must init every call)
  k_zero<<<dim3((PDIM + 255) / 256), cb, 0, stream>>>(usage, PDIM);

  // 1) bf16 conversions
  k_f32_to_bf16<<<cgrid, cb, 0, stream>>>(x, xbf, (size_t)MTOK * HDIM);
  k_f32_to_bf16<<<cgrid, cb, 0, stream>>>(w1, w1bf, (size_t)HDIM * HDIM);
  k_f32_to_bf16<<<cgrid, cb, 0, stream>>>(w2, w2bf, (size_t)PDIM * HDIM);
  k_f32_to_bf16<<<cgrid, cb, 0, stream>>>(wproj, wprojbf, (size_t)DDIM * HDIM);
  k_f32_to_bf16<<<cgrid, cb, 0, stream>>>(wmap, wmapbf, (size_t)DDIM * 2 * DDIM);
  k_f32_to_bf16<<<cgrid, cb, 0, stream>>>(wout, woutbf, (size_t)HDIM * H3);
  k_f32_to_bf16<<<cgrid, cb, 0, stream>>>(wrp, wrpbf, (size_t)HDIM * DDIM);
  k_convw_reorg<<<cgrid, cb, 0, stream>>>(conv_w, convwbf);

  // 2) local branch: 9 shifted GEMMs into convacc [M,3H], tap kk=0 carries bias
  const int dil[3] = {1, 2, 4};
  for (int d = 0; d < 3; ++d) {
    for (int kk = 0; kk < 3; ++kk) {
      int shift = (kk - 1) * dil[d];
      gemm_bf16_wmma<<<dim3(HDIM / BN, MTOK / BM), cb, 0, stream>>>(
          xbf, convwbf + (size_t)(d * 3 + kk) * HDIM * HDIM,
          kk == 0 ? (conv_b + d * HDIM) : nullptr,
          convacc + (size_t)d * HDIM,
          MTOK, HDIM, HDIM, H3, shift, kk == 0 ? 0 : GEMM_ACC);
    }
  }
  // 3) exact GELU -> bf16 concat activations
  k_gelu_bf16<<<cgrid, cb, 0, stream>>>(convacc, convact, (size_t)MTOK * H3);
  // 4) local_out = concat @ wout^T + bout   (overwrites dead convacc region)
  gemm_bf16_wmma<<<dim3(HDIM / BN, MTOK / BM), cb, 0, stream>>>(
      convact, woutbf, bout, localout, MTOK, HDIM, H3, HDIM, 0, 0);

  // 5) router MLP: inter = relu(x @ w1^T + b1) -> bf16
  gemm_bf16_wmma<<<dim3(HDIM / BN, MTOK / BM), cb, 0, stream>>>(
      xbf, w1bf, b1, interf, MTOK, HDIM, HDIM, HDIM, 0, GEMM_RELU);
  k_f32_to_bf16<<<cgrid, cb, 0, stream>>>(interf, interbf, (size_t)MTOK * HDIM);
  // 6) logits = inter @ w2^T + b2   (reuses region A)
  gemm_bf16_wmma<<<dim3(PDIM / BN, MTOK / BM), cb, 0, stream>>>(
      interbf, w2bf, b2, logits, MTOK, PDIM, HDIM, PDIM, 0, 0);
  // 7) top-2 + softmax + expert mix + usage
  k_router<<<dim3(MTOK), cb, 0, stream>>>(logits, pool, temp, wmapvec, usage);

  // 8) projected_x = LN(x @ wproj^T + bproj); pack with weighted_map as bf16
  gemm_bf16_wmma<<<dim3(DDIM / BN, MTOK / BM), cb, 0, stream>>>(
      xbf, wprojbf, bproj, projf, MTOK, DDIM, HDIM, DDIM, 0, 0);
  k_ln_pack<<<dim3(MTOK), cb, 0, stream>>>(projf, wmapvec, ln_g, ln_b, combbf);
  // 9) transformation = combined @ wmap^T + bmap
  gemm_bf16_wmma<<<dim3(DDIM / BN, MTOK / BM), cb, 0, stream>>>(
      combbf, wmapbf, bmap, transff, MTOK, DDIM, 2 * DDIM, DDIM, 0, 0);
  k_f32_to_bf16<<<cgrid, cb, 0, stream>>>(transff, transfbf, (size_t)MTOK * DDIM);
  // 10) global_out = transformation @ wrp^T + brp
  gemm_bf16_wmma<<<dim3(HDIM / BN, MTOK / BM), cb, 0, stream>>>(
      transfbf, wrpbf, brp, globalo, MTOK, HDIM, DDIM, HDIM, 0, 0);

  // 11) gate + combine -> d_out[0 : M*H]
  k_gate_combine<<<dim3(MTOK), cb, 0, stream>>>(x, wg, bg, localout, globalo, out);
  // 12) diversity loss -> d_out[M*H]
  k_divloss<<<dim3(1), cb, 0, stream>>>(usage, out + (size_t)MTOK * HDIM);
}
